// VisionGPT2Model_28424093565594
// MI455X (gfx1250) — compile-verified, matrix operand order checked by reading, not run on hardware
//
#include <hip/hip_runtime.h>
#include <math.h>

// ---------------- model dims ----------------
#define BDIM   4
#define DMODEL 768
#define NHEAD  12
#define HSIZE  64
#define DEPTH  6
#define VOCAB  50257
#define TDEC   512
#define TENC   197      // 196 patches + cls
#define NPATCH 196
#define MLPH   3072

typedef __attribute__((ext_vector_type(16))) _Float16 v16h;
typedef __attribute__((ext_vector_type(8)))  float    v8f;

union F16x16 { v16h v; unsigned long long q[4]; };

static __device__ __forceinline__ unsigned pack2h(float a, float b)
{
    union { _Float16 h[2]; unsigned u; } t;
    t.h[0] = (_Float16)a; t.h[1] = (_Float16)b;
    return t.u;
}

// =====================================================================
// Generic batched WMMA GEMM:  C = act(alpha * A @ B(^T) + bias) + resid
//   A: [M,K] row-major (lda), B: [K,N] (ldb) or [N,K] if transB, C: [M,N] (ldc)
//   per-z offsets: off = (z/zInner)*sXo + (z%zInner)*sXi   (X = A,B,C)
//   resid (if non-null) indexed identically to C.
// Block = 256 threads = 8 waves (2 x 4); block tile 64(M) x 128(N); K chunk 32.
// Each wave owns a 32x32 strip = 2x2 16x16 WMMA accumulators
// (each A / B fragment is reused twice -> 4 WMMAs per 4 fragment loads).
// LDS: A tile [m][k], B tile stored transposed [n][k]  -> contiguous frags.
// =====================================================================
#define BLK_M 64
#define BLK_N 128
#define BLK_K 32
#define TSTR  40   // LDS row stride in halves (80 B, 8B-aligned rows)

__global__ __launch_bounds__(256) void gemm_f16wmma(
    const float* __restrict__ A, const float* __restrict__ Bm,
    const float* __restrict__ bias, const float* __restrict__ resid,
    float* __restrict__ C,
    int M, int N, int K, int lda, int ldb, int ldc,
    int zInner,
    long long sAo, long long sAi, long long sBo, long long sBi,
    long long sCo, long long sCi,
    int transB, float alpha, int act)
{
    __shared__ _Float16 Ash[BLK_M * TSTR];   //  5.0 KB, [m][k]
    __shared__ _Float16 Bsh[BLK_N * TSTR];   // 10.0 KB, [n][k]

    const int tid  = threadIdx.x;
    const int lane = tid & 31;
    const int wave = tid >> 5;     // 0..7
    const int wm   = wave >> 2;    // 0..1  (32-row strip)
    const int wn   = wave & 3;     // 0..3  (32-col strip)

    const int z = blockIdx.z;
    const long long offA = (long long)(z / zInner) * sAo + (long long)(z % zInner) * sAi;
    const long long offB = (long long)(z / zInner) * sBo + (long long)(z % zInner) * sBi;
    const long long offC = (long long)(z / zInner) * sCo + (long long)(z % zInner) * sCi;

    const float* Ap = A  + offA;
    const float* Bp = Bm + offB;

    const int m0 = blockIdx.y * BLK_M;
    const int n0 = blockIdx.x * BLK_N;

    const bool alignedA = ((lda & 3) == 0);
    const bool alignedB = ((ldb & 3) == 0);
    const bool fullM = (m0 + BLK_M <= M);
    const bool fullN = (n0 + BLK_N <= N);

    v8f acc00 = {}, acc01 = {}, acc10 = {}, acc11 = {};

    for (int k0 = 0; k0 < K; k0 += BLK_K) {
        const bool fullK = (k0 + BLK_K <= K);

        // ---- stage A tile (64x32): two float4 tasks per thread ----
        {
            const bool fastA = fullK && fullM && alignedA;
#pragma unroll
            for (int t = 0; t < 2; ++t) {
                int task = tid + t * 256;          // 0..511
                int r  = task >> 3;                // 0..63
                int c4 = (task & 7) << 2;          // 0..28
                if (fastA) {
                    const float4 v = *(const float4*)(Ap + (long long)(m0 + r) * lda + k0 + c4);
                    unsigned* dst = (unsigned*)(Ash + r * TSTR + c4);
                    dst[0] = pack2h(v.x, v.y);
                    dst[1] = pack2h(v.z, v.w);
                } else {
#pragma unroll
                    for (int j = 0; j < 4; ++j) {
                        int gm = m0 + r, gk = k0 + c4 + j;
                        int cm = gm < M ? gm : M - 1;
                        int ck = gk < K ? gk : K - 1;
                        float v = Ap[(long long)cm * lda + ck];      // unconditional
                        v = (gm < M && gk < K) ? v : 0.0f;
                        Ash[r * TSTR + c4 + j] = (_Float16)v;
                    }
                }
            }
        }

        // ---- stage B tile (32k x 128n) into Bsh[n][k]: 4 float4 per thread ----
        if (transB) {
            const bool fastB = fullK && fullN && alignedB;
#pragma unroll
            for (int t = 0; t < 4; ++t) {
                int task = tid + t * 256;          // 0..1023
                int n  = task >> 3;                // 0..127
                int c4 = (task & 7) << 2;          // k offset
                if (fastB) {
                    const float4 v = *(const float4*)(Bp + (long long)(n0 + n) * ldb + k0 + c4);
                    unsigned* dst = (unsigned*)(Bsh + n * TSTR + c4);
                    dst[0] = pack2h(v.x, v.y);
                    dst[1] = pack2h(v.z, v.w);
                } else {
#pragma unroll
                    for (int j = 0; j < 4; ++j) {
                        int gn = n0 + n, gk = k0 + c4 + j;
                        int cn = gn < N ? gn : N - 1;
                        int ck = gk < K ? gk : K - 1;
                        float v = Bp[(long long)cn * ldb + ck];      // unconditional
                        v = (gn < N && gk < K) ? v : 0.0f;
                        Bsh[n * TSTR + c4 + j] = (_Float16)v;
                    }
                }
            }
        } else {
            const bool fastB = fullK && fullN && alignedB;
#pragma unroll
            for (int t = 0; t < 4; ++t) {
                int task = tid + t * 256;          // 0..1023
                int k  = task >> 5;                // 0..31
                int n4 = (task & 31) << 2;         // 0..124
                if (fastB) {
                    const float4 v = *(const float4*)(Bp + (long long)(k0 + k) * ldb + n0 + n4);
                    Bsh[(n4 + 0) * TSTR + k] = (_Float16)v.x;
                    Bsh[(n4 + 1) * TSTR + k] = (_Float16)v.y;
                    Bsh[(n4 + 2) * TSTR + k] = (_Float16)v.z;
                    Bsh[(n4 + 3) * TSTR + k] = (_Float16)v.w;
                } else {
#pragma unroll
                    for (int j = 0; j < 4; ++j) {
                        int gn = n0 + n4 + j, gk = k0 + k;
                        int cn = gn < N ? gn : N - 1;
                        int ck = gk < K ? gk : K - 1;
                        float v = Bp[(long long)ck * ldb + cn];      // unconditional
                        v = (gn < N && gk < K) ? v : 0.0f;
                        Bsh[(n4 + j) * TSTR + k] = (_Float16)v;
                    }
                }
            }
        }
        __syncthreads();

        // ---- fragments: contiguous 8-half runs -> b64/b128 DS loads ----
        const int kb  = (lane >> 4) * 8;          // 0 or 8
        const int ml0 = wm * 32 + (lane & 15);
        const int nl0 = wn * 32 + (lane & 15);

        F16x16 af0, af1, bf0, bf1;
        {
            const unsigned long long* r = (const unsigned long long*)(Ash + ml0 * TSTR);
            af0.q[0] = r[(kb >> 2) + 0];
            af0.q[1] = r[(kb >> 2) + 1];
            af0.q[2] = r[((16 + kb) >> 2) + 0];
            af0.q[3] = r[((16 + kb) >> 2) + 1];
        }
        {
            const unsigned long long* r = (const unsigned long long*)(Ash + (ml0 + 16) * TSTR);
            af1.q[0] = r[(kb >> 2) + 0];
            af1.q[1] = r[(kb >> 2) + 1];
            af1.q[2] = r[((16 + kb) >> 2) + 0];
            af1.q[3] = r[((16 + kb) >> 2) + 1];
        }
        {
            const unsigned long long* r = (const unsigned long long*)(Bsh + nl0 * TSTR);
            bf0.q[0] = r[(kb >> 2) + 0];
            bf0.q[1] = r[(kb >> 2) + 1];
            bf0.q[2] = r[((16 + kb) >> 2) + 0];
            bf0.q[3] = r[((16 + kb) >> 2) + 1];
        }
        {
            const unsigned long long* r = (const unsigned long long*)(Bsh + (nl0 + 16) * TSTR);
            bf1.q[0] = r[(kb >> 2) + 0];
            bf1.q[1] = r[(kb >> 2) + 1];
            bf1.q[2] = r[((16 + kb) >> 2) + 0];
            bf1.q[3] = r[((16 + kb) >> 2) + 1];
        }

        acc00 = __builtin_amdgcn_wmma_f32_16x16x32_f16(
                    false, af0.v, false, bf0.v, (short)0, acc00, false, false);
        acc01 = __builtin_amdgcn_wmma_f32_16x16x32_f16(
                    false, af0.v, false, bf1.v, (short)0, acc01, false, false);
        acc10 = __builtin_amdgcn_wmma_f32_16x16x32_f16(
                    false, af1.v, false, bf0.v, (short)0, acc10, false, false);
        acc11 = __builtin_amdgcn_wmma_f32_16x16x32_f16(
                    false, af1.v, false, bf1.v, (short)0, acc11, false, false);
        __syncthreads();
    }

    // ---- scatter C: lane n = l&15, VGPR r -> m = r + 8*(l>>4) ----
#pragma unroll
    for (int mi = 0; mi < 2; ++mi) {
#pragma unroll
        for (int ni = 0; ni < 2; ++ni) {
            const int gn = n0 + wn * 32 + ni * 16 + (lane & 15);
            if (gn >= N) continue;
            const float bv = bias ? bias[gn] : 0.0f;
            const v8f acc = mi ? (ni ? acc11 : acc10) : (ni ? acc01 : acc00);
#pragma unroll
            for (int r = 0; r < 8; ++r) {
                int gm = m0 + wm * 32 + mi * 16 + r + 8 * (lane >> 4);
                if (gm < M) {
                    long long ci = offC + (long long)gm * ldc + gn;
                    float v = acc[r] * alpha + bv;
                    if (act == 1) v = 0.5f * v * (1.0f + erff(v * 0.70710678118654752f));
                    if (resid) v += resid[ci];
                    C[ci] = v;
                }
            }
        }
    }
}

// ------------------- LayerNorm (one block per row) -------------------
__global__ __launch_bounds__(256) void layernorm_k(
    const float* __restrict__ x, const float* __restrict__ w,
    const float* __restrict__ b, float* __restrict__ o, int D)
{
    const long long row = blockIdx.x;
    const float* xr = x + row * D;
    float* orow = o + row * D;
    __shared__ float sm[8];
    const int tid = threadIdx.x, lane = tid & 31, wv = tid >> 5;

    float s = 0.0f;
    for (int i = tid; i < D; i += 256) s += xr[i];
    for (int m = 16; m >= 1; m >>= 1) s += __shfl_xor(s, m, 32);
    if (lane == 0) sm[wv] = s;
    __syncthreads();
    float tot = 0.0f;
    for (int i = 0; i < 8; ++i) tot += sm[i];
    const float mean = tot / (float)D;
    __syncthreads();

    float v = 0.0f;
    for (int i = tid; i < D; i += 256) { float d = xr[i] - mean; v += d * d; }
    for (int m = 16; m >= 1; m >>= 1) v += __shfl_xor(v, m, 32);
    if (lane == 0) sm[wv] = v;
    __syncthreads();
    float vtot = 0.0f;
    for (int i = 0; i < 8; ++i) vtot += sm[i];
    const float rstd = rsqrtf(vtot / (float)D + 1e-5f);

    for (int i = tid; i < D; i += 256)
        orow[i] = (xr[i] - mean) * rstd * w[i] + b[i];
}

// ------------- Softmax over score rows (one block per row) -----------
// row = z*Tq + q ; leading dim ld >= Tk ; masked (j >= lim) written 0.
__global__ __launch_bounds__(256) void softmax_k(
    float* __restrict__ S, int Tq, int Tk, int ld, int causal)
{
    const long long row = blockIdx.x;
    float* p = S + row * (long long)ld;
    const int q   = (int)(row % Tq);
    const int lim = causal ? (q + 1) : Tk;
    __shared__ float sm[8];
    const int tid = threadIdx.x, lane = tid & 31, wv = tid >> 5;

    float mx = -3.4e38f;
    for (int j = tid; j < lim; j += 256) mx = fmaxf(mx, p[j]);
    for (int m = 16; m >= 1; m >>= 1) mx = fmaxf(mx, __shfl_xor(mx, m, 32));
    if (lane == 0) sm[wv] = mx;
    __syncthreads();
    float rmax = -3.4e38f;
    for (int i = 0; i < 8; ++i) rmax = fmaxf(rmax, sm[i]);
    __syncthreads();

    float sum = 0.0f;
    for (int j = tid; j < lim; j += 256) {
        float e = __expf(p[j] - rmax);
        p[j] = e;
        sum += e;
    }
    for (int m = 16; m >= 1; m >>= 1) sum += __shfl_xor(sum, m, 32);
    if (lane == 0) sm[wv] = sum;
    __syncthreads();
    float tot = 0.0f;
    for (int i = 0; i < 8; ++i) tot += sm[i];
    const float inv = 1.0f / tot;
    __syncthreads();

    for (int j = tid; j < Tk; j += 256)
        p[j] = (j < lim) ? p[j] * inv : 0.0f;
}

// ---------- patch unfold: image[B,3,224,224] -> xp[B,196,768] --------
__global__ void patchify_k(const float* __restrict__ img, float* __restrict__ xp)
{
    const long long n = (long long)BDIM * NPATCH * DMODEL;
    for (long long i = (long long)blockIdx.x * blockDim.x + threadIdx.x;
         i < n; i += (long long)gridDim.x * blockDim.x) {
        int j  = (int)(i % DMODEL);
        long long bp = i / DMODEL;
        int p  = (int)(bp % NPATCH);
        int b  = (int)(bp / NPATCH);
        int c  = j >> 8;
        int rr = j & 255;
        int py = rr >> 4, px = rr & 15;
        int gy = p / 14, gx = p % 14;
        long long src = (((long long)b * 3 + c) * 224 + (gy * 16 + py)) * 224
                        + (gx * 16 + px);
        xp[i] = img[src];
    }
}

// ---- xe[b,0,:] = cls + pos[0];  xe[b,t>0,:] += pos[t]  --------------
__global__ void clspos_k(float* __restrict__ xe, const float* __restrict__ cls,
                         const float* __restrict__ pos)
{
    const long long n = (long long)BDIM * TENC * DMODEL;
    for (long long i = (long long)blockIdx.x * blockDim.x + threadIdx.x;
         i < n; i += (long long)gridDim.x * blockDim.x) {
        int d = (int)(i % DMODEL);
        long long bt = i / DMODEL;
        int t = (int)(bt % TENC);
        float base = (t == 0) ? cls[d] : xe[i];
        xe[i] = base + pos[(long long)t * DMODEL + d];
    }
}

// ---- y[b,t,:] = wte[ids[b,t],:] + wpe[t,:] --------------------------
__global__ void embed_k(const int* __restrict__ ids, const float* __restrict__ wte,
                        const float* __restrict__ wpe, float* __restrict__ y)
{
    const long long n = (long long)BDIM * TDEC * DMODEL;
    for (long long i = (long long)blockIdx.x * blockDim.x + threadIdx.x;
         i < n; i += (long long)gridDim.x * blockDim.x) {
        int d = (int)(i % DMODEL);
        long long bt = i / DMODEL;
        int t = (int)(bt % TDEC);
        int tok = ids[bt];
        y[i] = wte[(long long)tok * DMODEL + d] + wpe[(long long)t * DMODEL + d];
    }
}

// =====================================================================
// host side
// =====================================================================
static inline void gemm(hipStream_t s,
    const float* A, const float* B, const float* bias, const float* resid, float* C,
    int M, int N, int K, int lda, int ldb, int ldc,
    int nz, int zInner,
    long long sAo, long long sAi, long long sBo, long long sBi,
    long long sCo, long long sCi,
    int transB, float alpha, int act)
{
    dim3 g((N + BLK_N - 1) / BLK_N, (M + BLK_M - 1) / BLK_M, nz);
    gemm_f16wmma<<<g, dim3(256), 0, s>>>(A, B, bias, resid, C, M, N, K,
        lda, ldb, ldc, zInner, sAo, sAi, sBo, sBi, sCo, sCi, transB, alpha, act);
}

extern "C" void kernel_launch(void* const* d_in, const int* in_sizes, int n_in,
                              void* d_out, int out_size, void* d_ws, size_t ws_size,
                              hipStream_t stream)
{
    (void)in_sizes; (void)n_in; (void)out_size; (void)ws_size;
    const float* image   = (const float*)d_in[0];
    const int*   ids     = (const int*)  d_in[1];
    const float* patch_w = (const float*)d_in[2];
    const float* patch_b = (const float*)d_in[3];
    const float* clsp    = (const float*)d_in[4];
    const float* pos     = (const float*)d_in[5];
    const float* v_ln1w = (const float*)d_in[6];
    const float* v_ln1b = (const float*)d_in[7];
    const float* v_qkvw = (const float*)d_in[8];
    const float* v_qkvb = (const float*)d_in[9];
    const float* v_prjw = (const float*)d_in[10];
    const float* v_prjb = (const float*)d_in[11];
    const float* v_ln2w = (const float*)d_in[12];
    const float* v_ln2b = (const float*)d_in[13];
    const float* v_fc1w = (const float*)d_in[14];
    const float* v_fc1b = (const float*)d_in[15];
    const float* v_fc2w = (const float*)d_in[16];
    const float* v_fc2b = (const float*)d_in[17];
    const float* wte    = (const float*)d_in[18];
    const float* wpe    = (const float*)d_in[19];
    const float* d_ln1w = (const float*)d_in[20];
    const float* d_ln1b = (const float*)d_in[21];
    const float* d_atw  = (const float*)d_in[22];
    const float* d_atb  = (const float*)d_in[23];
    const float* d_apw  = (const float*)d_in[24];
    const float* d_apb  = (const float*)d_in[25];
    const float* d_ln2w = (const float*)d_in[26];
    const float* d_ln2b = (const float*)d_in[27];
    const float* d_qw   = (const float*)d_in[28];
    const float* d_qb   = (const float*)d_in[29];
    const float* d_kw   = (const float*)d_in[30];
    const float* d_kb   = (const float*)d_in[31];
    const float* d_vw   = (const float*)d_in[32];
    const float* d_vb   = (const float*)d_in[33];
    const float* d_cpw  = (const float*)d_in[34];
    const float* d_cpb  = (const float*)d_in[35];
    const float* d_ln3w = (const float*)d_in[36];
    const float* d_ln3b = (const float*)d_in[37];
    const float* d_fcw  = (const float*)d_in[38];
    const float* d_fcb  = (const float*)d_in[39];
    const float* d_fpw  = (const float*)d_in[40];
    const float* d_fpb  = (const float*)d_in[41];
    const float* lnf_w  = (const float*)d_in[42];
    const float* lnf_b  = (const float*)d_in[43];

    // -------- workspace layout (floats) --------
    float* ws = (float*)d_ws;
    size_t off = 0;
    float* xe   = ws + off; off += (size_t)BDIM * TENC * DMODEL;
    float* hbuf = ws + off; off += (size_t)BDIM * TDEC * DMODEL;
    float* qkvb = ws + off; off += (size_t)BDIM * TDEC * 3 * DMODEL;
    float* scr  = ws + off; off += (size_t)BDIM * NHEAD * TDEC * TDEC;
    float* att  = ws + off; off += (size_t)BDIM * TDEC * DMODEL;
    float* mlph = ws + off; off += (size_t)BDIM * TDEC * MLPH;
    float* ybuf = ws + off; off += (size_t)BDIM * TDEC * DMODEL;
    float* qcb  = ws + off; off += (size_t)BDIM * TDEC * DMODEL;
    float* kcb  = ws + off; off += (size_t)BDIM * TENC * DMODEL;
    float* vcb  = ws + off; off += (size_t)BDIM * TENC * DMODEL;

    const int encRows = BDIM * TENC;   // 788
    const int decRows = BDIM * TDEC;   // 2048
    const float iss = 0.125f;          // 1/sqrt(64)
    const int ldE = 200;               // padded score stride for Tk=197

    // ---------------- ViT patch embed ----------------
    patchify_k<<<1024, 256, 0, stream>>>(image, mlph);
    gemm(stream, mlph, patch_w, patch_b, nullptr, xe + DMODEL,
         NPATCH, DMODEL, DMODEL, DMODEL, DMODEL, DMODEL,
         BDIM, 1, (long long)NPATCH * DMODEL, 0, 0, 0,
         (long long)TENC * DMODEL, 0, 0, 1.0f, 0);
    clspos_k<<<1024, 256, 0, stream>>>(xe, clsp, pos);

    // ---------------- ViT blocks ----------------
    for (int i = 0; i < DEPTH; ++i) {
        layernorm_k<<<encRows, 256, 0, stream>>>(xe, v_ln1w + i*DMODEL, v_ln1b + i*DMODEL, hbuf, DMODEL);
        gemm(stream, hbuf, v_qkvw + (size_t)i*DMODEL*3*DMODEL, v_qkvb + (size_t)i*3*DMODEL,
             nullptr, qkvb, encRows, 3*DMODEL, DMODEL, DMODEL, 3*DMODEL, 3*DMODEL,
             1, 1, 0,0,0,0,0,0, 0, 1.0f, 0);
        // scores = Q K^T / 8   (z = b*H + h), ldc padded to ldE
        gemm(stream, qkvb, qkvb + DMODEL, nullptr, nullptr, scr,
             TENC, TENC, HSIZE, 3*DMODEL, 3*DMODEL, ldE,
             BDIM*NHEAD, NHEAD,
             (long long)TENC*3*DMODEL, HSIZE,
             (long long)TENC*3*DMODEL, HSIZE,
             (long long)NHEAD*TENC*ldE, (long long)TENC*ldE,
             1, iss, 0);
        softmax_k<<<BDIM*NHEAD*TENC, 256, 0, stream>>>(scr, TENC, TENC, ldE, 0);
        // O = P V
        gemm(stream, scr, qkvb + 2*DMODEL, nullptr, nullptr, att,
             TENC, HSIZE, TENC, ldE, 3*DMODEL, DMODEL,
             BDIM*NHEAD, NHEAD,
             (long long)NHEAD*TENC*ldE, (long long)TENC*ldE,
             (long long)TENC*3*DMODEL, HSIZE,
             (long long)TENC*DMODEL, HSIZE,
             0, 1.0f, 0);
        gemm(stream, att, v_prjw + (size_t)i*DMODEL*DMODEL, v_prjb + (size_t)i*DMODEL,
             xe, xe, encRows, DMODEL, DMODEL, DMODEL, DMODEL, DMODEL,
             1, 1, 0,0,0,0,0,0, 0, 1.0f, 0);
        layernorm_k<<<encRows, 256, 0, stream>>>(xe, v_ln2w + i*DMODEL, v_ln2b + i*DMODEL, hbuf, DMODEL);
        gemm(stream, hbuf, v_fc1w + (size_t)i*DMODEL*MLPH, v_fc1b + (size_t)i*MLPH,
             nullptr, mlph, encRows, MLPH, DMODEL, DMODEL, MLPH, MLPH,
             1, 1, 0,0,0,0,0,0, 0, 1.0f, 1 /*gelu*/);
        gemm(stream, mlph, v_fc2w + (size_t)i*MLPH*DMODEL, v_fc2b + (size_t)i*DMODEL,
             xe, xe, encRows, DMODEL, MLPH, MLPH, DMODEL, DMODEL,
             1, 1, 0,0,0,0,0,0, 0, 1.0f, 0);
    }

    // ---------------- decoder embed ----------------
    embed_k<<<1024, 256, 0, stream>>>(ids, wte, wpe, ybuf);

    // ---------------- GPT2 blocks ----------------
    for (int i = 0; i < DEPTH; ++i) {
        // causal self-attention
        layernorm_k<<<decRows, 256, 0, stream>>>(ybuf, d_ln1w + i*DMODEL, d_ln1b + i*DMODEL, hbuf, DMODEL);
        gemm(stream, hbuf, d_atw + (size_t)i*DMODEL*3*DMODEL, d_atb + (size_t)i*3*DMODEL,
             nullptr, qkvb, decRows, 3*DMODEL, DMODEL, DMODEL, 3*DMODEL, 3*DMODEL,
             1, 1, 0,0,0,0,0,0, 0, 1.0f, 0);
        gemm(stream, qkvb, qkvb + DMODEL, nullptr, nullptr, scr,
             TDEC, TDEC, HSIZE, 3*DMODEL, 3*DMODEL, TDEC,
             BDIM*NHEAD, NHEAD,
             (long long)TDEC*3*DMODEL, HSIZE,
             (long long)TDEC*3*DMODEL, HSIZE,
             (long long)NHEAD*TDEC*TDEC, (long long)TDEC*TDEC,
             1, iss, 0);
        softmax_k<<<BDIM*NHEAD*TDEC, 256, 0, stream>>>(scr, TDEC, TDEC, TDEC, 1);
        gemm(stream, scr, qkvb + 2*DMODEL, nullptr, nullptr, att,
             TDEC, HSIZE, TDEC, TDEC, 3*DMODEL, DMODEL,
             BDIM*NHEAD, NHEAD,
             (long long)NHEAD*TDEC*TDEC, (long long)TDEC*TDEC,
             (long long)TDEC*3*DMODEL, HSIZE,
             (long long)TDEC*DMODEL, HSIZE,
             0, 1.0f, 0);
        gemm(stream, att, d_apw + (size_t)i*DMODEL*DMODEL, d_apb + (size_t)i*DMODEL,
             ybuf, ybuf, decRows, DMODEL, DMODEL, DMODEL, DMODEL, DMODEL,
             1, 1, 0,0,0,0,0,0, 0, 1.0f, 0);

        // cross-attention
        layernorm_k<<<decRows, 256, 0, stream>>>(ybuf, d_ln2w + i*DMODEL, d_ln2b + i*DMODEL, hbuf, DMODEL);
        gemm(stream, hbuf, d_qw + (size_t)i*DMODEL*DMODEL, d_qb + (size_t)i*DMODEL,
             nullptr, qcb, decRows, DMODEL, DMODEL, DMODEL, DMODEL, DMODEL,
             1, 1, 0,0,0,0,0,0, 0, 1.0f, 0);
        gemm(stream, xe, d_kw + (size_t)i*DMODEL*DMODEL, d_kb + (size_t)i*DMODEL,
             nullptr, kcb, encRows, DMODEL, DMODEL, DMODEL, DMODEL, DMODEL,
             1, 1, 0,0,0,0,0,0, 0, 1.0f, 0);
        gemm(stream, xe, d_vw + (size_t)i*DMODEL*DMODEL, d_vb + (size_t)i*DMODEL,
             nullptr, vcb, encRows, DMODEL, DMODEL, DMODEL, DMODEL, DMODEL,
             1, 1, 0,0,0,0,0,0, 0, 1.0f, 0);
        gemm(stream, qcb, kcb, nullptr, nullptr, scr,
             TDEC, TENC, HSIZE, DMODEL, DMODEL, ldE,
             BDIM*NHEAD, NHEAD,
             (long long)TDEC*DMODEL, HSIZE,
             (long long)TENC*DMODEL, HSIZE,
             (long long)NHEAD*TDEC*ldE, (long long)TDEC*ldE,
             1, iss, 0);
        softmax_k<<<BDIM*NHEAD*TDEC, 256, 0, stream>>>(scr, TDEC, TENC, ldE, 0);
        gemm(stream, scr, vcb, nullptr, nullptr, att,
             TDEC, HSIZE, TENC, ldE, DMODEL, DMODEL,
             BDIM*NHEAD, NHEAD,
             (long long)NHEAD*TDEC*ldE, (long long)TDEC*ldE,
             (long long)TENC*DMODEL, HSIZE,
             (long long)TDEC*DMODEL, HSIZE,
             0, 1.0f, 0);
        gemm(stream, att, d_cpw + (size_t)i*DMODEL*DMODEL, d_cpb + (size_t)i*DMODEL,
             ybuf, ybuf, decRows, DMODEL, DMODEL, DMODEL, DMODEL, DMODEL,
             1, 1, 0,0,0,0,0,0, 0, 1.0f, 0);

        // MLP
        layernorm_k<<<decRows, 256, 0, stream>>>(ybuf, d_ln3w + i*DMODEL, d_ln3b + i*DMODEL, hbuf, DMODEL);
        gemm(stream, hbuf, d_fcw + (size_t)i*DMODEL*MLPH, d_fcb + (size_t)i*MLPH,
             nullptr, mlph, decRows, MLPH, DMODEL, DMODEL, MLPH, MLPH,
             1, 1, 0,0,0,0,0,0, 0, 1.0f, 1 /*gelu*/);
        gemm(stream, mlph, d_fpw + (size_t)i*MLPH*DMODEL, d_fpb + (size_t)i*DMODEL,
             ybuf, ybuf, decRows, DMODEL, MLPH, MLPH, DMODEL, DMODEL,
             1, 1, 0,0,0,0,0,0, 0, 1.0f, 0);
    }

    // ---------------- final LN + tied LM head ----------------
    layernorm_k<<<decRows, 256, 0, stream>>>(ybuf, lnf_w, lnf_b, hbuf, DMODEL);
    gemm(stream, hbuf, wte, nullptr, nullptr, (float*)d_out,
         decRows, VOCAB, DMODEL, DMODEL, DMODEL, VOCAB,
         1, 1, 0,0,0,0,0,0, 1 /*B^T*/, 1.0f, 0);
}
